// GCN_14173392077062
// MI455X (gfx1250) — matrix operand order repaired and checked by reading, not compiled, not run on hardware
//
#include <hip/hip_runtime.h>
#include <hip/hip_bf16.h>

// ---------------- problem constants (match reference) ----------------
#define NN    50000      // nodes
#define NE    800000     // edges
#define F_IN  128
#define F_HID 256
#define F_OUT 40
#define F_OUT_PAD 48     // padded to 3 WMMA n-tiles
#define BN_EPS 1e-5f

typedef __bf16 bf16_t;
typedef __attribute__((ext_vector_type(16))) __bf16 v16bf;
typedef __attribute__((ext_vector_type(8)))  __bf16 v8bf;
typedef __attribute__((ext_vector_type(8)))  float  v8f;
typedef __attribute__((ext_vector_type(4)))  float  v4f;

// ---------------- utility kernels ----------------
__global__ void k_zero(float* __restrict__ p, long long n) {
  long long i = (long long)blockIdx.x * blockDim.x + threadIdx.x;
  long long s = (long long)gridDim.x * blockDim.x;
  for (; i < n; i += s) p[i] = 0.0f;
}

__global__ void k_degree(const int* __restrict__ src, const int* __restrict__ dst,
                         float* __restrict__ odeg, float* __restrict__ ideg, int e) {
  int i = blockIdx.x * blockDim.x + threadIdx.x;
  if (i < e) {
    atomicAdd(&odeg[src[i]], 1.0f);
    atomicAdd(&ideg[dst[i]], 1.0f);
  }
}

__global__ void k_norm(float* __restrict__ a, float* __restrict__ b, int n) {
  int i = blockIdx.x * blockDim.x + threadIdx.x;
  if (i < n) {
    a[i] = rsqrtf(fmaxf(a[i], 1.0f));
    b[i] = rsqrtf(fmaxf(b[i], 1.0f));
  }
}

__global__ void k_cvt_bf16(const float* __restrict__ x, bf16_t* __restrict__ y, long long n) {
  long long i = (long long)blockIdx.x * blockDim.x + threadIdx.x;
  long long s = (long long)gridDim.x * blockDim.x;
  for (; i < n; i += s) y[i] = (bf16_t)x[i];
}

// Pack weight W[K][Nv] (row-major f32) into per-wave WMMA B fragments (bf16):
//   P[((kt*NT + nt)*32 + lane)*16 + e] = W[kt*32 + 16*(lane/16) + e][nt*16 + lane%16]
// (contiguous k order, same as the A loads: lanes 0-15 hold K 0..15, lanes 16-31 hold K 16..31)
__global__ void k_packW(const float* __restrict__ W, bf16_t* __restrict__ P,
                        int K, int Nv, int NT) {
  long long total = (long long)(K / 32) * NT * 512;
  long long i = (long long)blockIdx.x * blockDim.x + threadIdx.x;
  long long s = (long long)gridDim.x * blockDim.x;
  for (; i < total; i += s) {
    int e    = (int)(i & 15);
    int lane = (int)((i >> 4) & 31);
    int nt   = (int)((i >> 9) % NT);
    int kt   = (int)(i / ((long long)512 * NT));
    int k = kt * 32 + (lane >> 4) * 16 + e;
    int n = nt * 16 + (lane & 15);
    float v = (n < Nv) ? W[(size_t)k * Nv + n] : 0.0f;
    P[i] = (bf16_t)v;
  }
}

// ---------------- WMMA GEMM: H[m][n] = sum_k A[m][k]*B[k][n], scaled by srcnorm[m] --------------
// One wave per 16x(16*NB) output tile: A fragment loaded once per K-step, reused by NB
// independent v_wmma_f32_16x16x32_bf16 accumulator chains (NB=4 for N=256, NB=3 for N=48).
// All fragment loads of a K-step are issued before the WMMA chain so the scheduler can use
// partial s_wait_loadcnt and overlap remaining loads with matrix ops.
template<int NB>
__global__ void __launch_bounds__(256)
k_gemm_wmma(const bf16_t* __restrict__ A, int lda,
            const bf16_t* __restrict__ Bp,
            const float* __restrict__ srcnorm,
            float* __restrict__ H, int ldh,
            int Mtiles, int NT, int KT) {
  int wave = (blockIdx.x * blockDim.x + threadIdx.x) >> 5;
  int lane = threadIdx.x & 31;
  int groups = NT / NB;
  if (wave >= Mtiles * groups) return;      // wave-uniform: EXEC stays all-ones
  int tm  = wave / groups;
  int tn0 = (wave % groups) * NB;
  int m0 = tm * 16;
  int row = m0 + (lane & 15);
  int khalf = lane >> 4;

  v8f c[NB];
#pragma unroll
  for (int j = 0; j < NB; ++j) c[j] = (v8f){};

  const bf16_t* arow = A + (size_t)row * lda + 16 * khalf;
  for (int kt = 0; kt < KT; ++kt) {
    // Issue ALL loads of this K-step first (A: two 16B chunks, B: NB x 32B), then the WMMAs.
    v8bf alo = *(const v8bf*)(arow + kt * 32);
    v8bf ahi = *(const v8bf*)(arow + kt * 32 + 8);
    const bf16_t* brow = Bp + (((size_t)kt * NT + tn0) * 32 + lane) * 16;
    v16bf b[NB];
#pragma unroll
    for (int j = 0; j < NB; ++j)
      b[j] = *(const v16bf*)(brow + (size_t)j * 512);   // n-tiles contiguous (512 elems apart)

    v16bf a;
#pragma unroll
    for (int i = 0; i < 8; ++i) { a[i] = alo[i]; a[8 + i] = ahi[i]; }
#pragma unroll
    for (int j = 0; j < NB; ++j) {
      c[j] = __builtin_amdgcn_wmma_f32_16x16x32_bf16(
          /*neg_a=*/false, a, /*neg_b=*/false, b[j],
          /*c_mod=*/(short)0, c[j], /*reuse_a=*/false, /*reuse_b=*/false);
    }
  }

  // D layout: row = m0 + 8*khalf + r, col = tn*16 + lane%16 ; fuse src_norm scale
  int colb = tn0 * 16 + (lane & 15);
  int rbase = m0 + 8 * khalf;
#pragma unroll
  for (int r = 0; r < 8; ++r) {
    int rr = rbase + r;
    float sn = srcnorm[rr];
    float* hrow = H + (size_t)rr * ldh + colb;
#pragma unroll
    for (int j = 0; j < NB; ++j) hrow[j * 16] = c[j][r] * sn;
  }
}

// ---------------- SpMM scatter: agg[dst] += H[src] (edge-parallel, float4 granularity) ---------
__global__ void k_spmm(const float* __restrict__ H, const int* __restrict__ src,
                       const int* __restrict__ dst, float* __restrict__ agg,
                       int nedges, int f4, int ld) {
  long long total = (long long)nedges * f4;
  long long i = (long long)blockIdx.x * blockDim.x + threadIdx.x;
  long long s = (long long)gridDim.x * blockDim.x;
  for (; i < total; i += s) {
    int e = (int)(i / f4);
    int c = (int)(i - (long long)e * f4) * 4;
    int sn = src[e], dn = dst[e];
    v4f v = *(const v4f*)(H + (size_t)sn * ld + c);
    float* p = agg + (size_t)dn * ld + c;
    atomicAdd(p + 0, v[0]);
    atomicAdd(p + 1, v[1]);
    atomicAdd(p + 2, v[2]);
    atomicAdd(p + 3, v[3]);
  }
}

// ---------------- finalize: h = agg*dst_norm + b (+res); accumulate BN channel sums ------------
__global__ void __launch_bounds__(256)
k_finalize(const float* __restrict__ agg, const float* __restrict__ dnorm,
           const float* __restrict__ bias, const float* __restrict__ res,
           float* __restrict__ h, float* __restrict__ sum, float* __restrict__ sumsq,
           int nodes, int chunk) {
  int c = threadIdx.x;                       // 256 channels
  int n0 = blockIdx.x * chunk;
  int n1 = n0 + chunk; if (n1 > nodes) n1 = nodes;
  float s = 0.0f, s2 = 0.0f;
  float bc = bias[c];
  for (int n = n0; n < n1; ++n) {
    size_t off = (size_t)n * 256 + c;
    float v = agg[off] * dnorm[n] + bc;
    if (res) v += res[off];
    h[off] = v;
    s += v; s2 += v * v;
  }
  atomicAdd(&sum[c], s);
  atomicAdd(&sumsq[c], s2);
}

__global__ void k_bnstats(const float* __restrict__ sum, const float* __restrict__ sumsq,
                          const float* __restrict__ g, const float* __restrict__ be,
                          float* __restrict__ scale, float* __restrict__ shift, float inv_n) {
  int c = threadIdx.x;
  float m = sum[c] * inv_n;
  float var = sumsq[c] * inv_n - m * m;
  float is = rsqrtf(var + BN_EPS);
  float sc = g[c] * is;
  scale[c] = sc;
  shift[c] = be[c] - m * sc;
}

__global__ void k_bnapply(const float* __restrict__ h, const float* __restrict__ scale,
                          const float* __restrict__ shift, float* __restrict__ hp,
                          bf16_t* __restrict__ hb, long long n) {
  long long i = (long long)blockIdx.x * blockDim.x + threadIdx.x;
  long long s = (long long)gridDim.x * blockDim.x;
  for (; i < n; i += s) {
    int c = (int)(i & 255);
    float v = fmaxf(h[i] * scale[c] + shift[c], 0.0f);
    hp[i] = v;
    hb[i] = (bf16_t)v;
  }
}

// ---------------- layer-2 output: out[n][c] = agg[n][c]*dst_norm[n] + b2[c], c<40 --------------
__global__ void k_out(const float* __restrict__ agg, const float* __restrict__ dnorm,
                      const float* __restrict__ bias, float* __restrict__ out) {
  long long total = (long long)NN * F_OUT;
  long long i = (long long)blockIdx.x * blockDim.x + threadIdx.x;
  long long s = (long long)gridDim.x * blockDim.x;
  for (; i < total; i += s) {
    int n = (int)(i / F_OUT);
    int c = (int)(i - (long long)n * F_OUT);
    out[i] = agg[(size_t)n * F_OUT_PAD + c] * dnorm[n] + bias[c];
  }
}

extern "C" void kernel_launch(void* const* d_in, const int* in_sizes, int n_in,
                              void* d_out, int out_size, void* d_ws, size_t ws_size,
                              hipStream_t stream) {
  const float* feat = (const float*)d_in[0];
  const int*   src  = (const int*)d_in[1];
  const int*   dst  = (const int*)d_in[2];
  const float* W0 = (const float*)d_in[3];
  const float* b0 = (const float*)d_in[4];
  const float* W1 = (const float*)d_in[5];
  const float* b1 = (const float*)d_in[6];
  const float* W2 = (const float*)d_in[7];
  const float* b2 = (const float*)d_in[8];
  const float* g0 = (const float*)d_in[9];
  const float* be0 = (const float*)d_in[10];
  const float* g1 = (const float*)d_in[11];
  const float* be1 = (const float*)d_in[12];
  float* out = (float*)d_out;
  (void)in_sizes; (void)n_in; (void)out_size; (void)ws_size;

  // ---- workspace carve-out (~193 MB) ----
  char* ws = (char*)d_ws;
  size_t o = 0;
  auto take = [&](size_t bytes) -> char* {
    char* p = ws + o;
    o += (bytes + 255) & ~(size_t)255;
    return p;
  };
  float*  snorm = (float*)take((size_t)NN * 4);
  float*  dnorm = (float*)take((size_t)NN * 4);
  bf16_t* featb = (bf16_t*)take((size_t)NN * F_IN * 2);
  bf16_t* w0p   = (bf16_t*)take((size_t)(F_IN / 32) * 16 * 512 * 2);
  bf16_t* w1p   = (bf16_t*)take((size_t)(F_HID / 32) * 16 * 512 * 2);
  bf16_t* w2p   = (bf16_t*)take((size_t)(F_HID / 32) * 3 * 512 * 2);
  float*  H     = (float*)take((size_t)NN * F_HID * 4);   // GEMM out / pre-BN
  float*  AGG   = (float*)take((size_t)NN * F_HID * 4);   // scatter target
  float*  HP    = (float*)take((size_t)NN * F_HID * 4);   // post-ReLU f32 (residual)
  bf16_t* HB    = (bf16_t*)take((size_t)NN * F_HID * 2);  // post-ReLU bf16 (GEMM A)
  float*  sum   = (float*)take(256 * 4);
  float*  sumsq = (float*)take(256 * 4);
  float*  scale = (float*)take(256 * 4);
  float*  shift = (float*)take(256 * 4);

  const int B = 256;
  const int MT = NN / 16;  // 3125 m-tiles (exact)
  const float inv_n = 1.0f / (float)NN;

  // ---- degrees + norms ----
  k_zero<<<256, B, 0, stream>>>(snorm, NN);
  k_zero<<<256, B, 0, stream>>>(dnorm, NN);
  k_degree<<<(NE + B - 1) / B, B, 0, stream>>>(src, dst, snorm, dnorm, NE);
  k_norm<<<(NN + B - 1) / B, B, 0, stream>>>(snorm, dnorm, NN);

  // ---- one-time bf16 conversion + weight packing ----
  k_cvt_bf16<<<(int)(((long long)NN * F_IN + B - 1) / B), B, 0, stream>>>(feat, featb, (long long)NN * F_IN);
  k_packW<<<128, B, 0, stream>>>(W0, w0p, F_IN, F_HID, 16);
  k_packW<<<256, B, 0, stream>>>(W1, w1p, F_HID, F_HID, 16);
  k_packW<<<48,  B, 0, stream>>>(W2, w2p, F_HID, F_OUT, 3);

  // waves per GEMM launch: Mtiles * (NT/NB); 8 waves per 256-thread block
  const int waves01 = MT * (16 / 4);   // 12500
  const int waves2  = MT * (3 / 3);    // 3125

  // ================= layer 0: 128 -> 256 =================
  k_gemm_wmma<4><<<(waves01 + 7) / 8, B, 0, stream>>>(featb, F_IN, w0p, snorm, H, F_HID, MT, 16, F_IN / 32);
  k_zero<<<50000, B, 0, stream>>>(AGG, (long long)NN * F_HID);
  k_spmm<<<(int)(((long long)NE * 64 + B - 1) / B), B, 0, stream>>>(H, src, dst, AGG, NE, 64, F_HID);
  k_zero<<<1, B, 0, stream>>>(sum, 256);
  k_zero<<<1, B, 0, stream>>>(sumsq, 256);
  k_finalize<<<(NN + 31) / 32, B, 0, stream>>>(AGG, dnorm, b0, nullptr, H, sum, sumsq, NN, 32);
  k_bnstats<<<1, B, 0, stream>>>(sum, sumsq, g0, be0, scale, shift, inv_n);
  k_bnapply<<<50000, B, 0, stream>>>(H, scale, shift, HP, HB, (long long)NN * F_HID);

  // ================= layer 1: 256 -> 256 (+residual) =================
  k_gemm_wmma<4><<<(waves01 + 7) / 8, B, 0, stream>>>(HB, F_HID, w1p, snorm, H, F_HID, MT, 16, F_HID / 32);
  k_zero<<<50000, B, 0, stream>>>(AGG, (long long)NN * F_HID);
  k_spmm<<<(int)(((long long)NE * 64 + B - 1) / B), B, 0, stream>>>(H, src, dst, AGG, NE, 64, F_HID);
  k_zero<<<1, B, 0, stream>>>(sum, 256);
  k_zero<<<1, B, 0, stream>>>(sumsq, 256);
  k_finalize<<<(NN + 31) / 32, B, 0, stream>>>(AGG, dnorm, b1, HP, H, sum, sumsq, NN, 32);
  k_bnstats<<<1, B, 0, stream>>>(sum, sumsq, g1, be1, scale, shift, inv_n);
  k_bnapply<<<50000, B, 0, stream>>>(H, scale, shift, HP, HB, (long long)NN * F_HID);

  // ================= layer 2: 256 -> 40 (padded 48) =================
  k_gemm_wmma<3><<<(waves2 + 7) / 8, B, 0, stream>>>(HB, F_HID, w2p, snorm, H, F_OUT_PAD, MT, 3, F_HID / 32);
  k_zero<<<10000, B, 0, stream>>>(AGG, (long long)NN * F_OUT_PAD);
  k_spmm<<<(int)(((long long)NE * 12 + B - 1) / B), B, 0, stream>>>(H, src, dst, AGG, NE, 12, F_OUT_PAD);
  k_out<<<(int)(((long long)NN * F_OUT + B - 1) / B), B, 0, stream>>>(AGG, dnorm, b2, out);
}
